// Jastrow_48627619726079
// MI455X (gfx1250) — matrix operand order repaired and checked by reading, not compiled, not run on hardware
//
#include <hip/hip_runtime.h>
#include <math.h>

// MI455X / gfx1250, wave32. One 288-thread workgroup (9 waves) per batch
// element: waves 0-7 -> the 8 psi pair-tiles (120 pairs padded to 128),
// wave 8 -> the phi particle tile. All three MLP layers run on
// V_WMMA_F32_16X16X32_F16 (f32 accumulate); GELU is exact-erf via a
// branchless A&S 7.1.26 rational approx (|err| < 1.5e-7, below the f16
// operand quantization noise).

typedef __attribute__((ext_vector_type(16))) _Float16 v16h;
typedef __attribute__((ext_vector_type(2)))  _Float16 h2;
typedef __attribute__((ext_vector_type(8)))  float    v8f;

union Frag { v16h v; h2 p[8]; };

__device__ __forceinline__ float gelu_exact(float x) {
    // 0.5*x*(1+erf(x/sqrt(2))) -- branchless: rcp + exp + 6 fma, no exec masks
    const float z    = x * 0.70710678118654752440f;
    const float az   = __builtin_fabsf(z);
    const float t    = __builtin_amdgcn_rcpf(__builtin_fmaf(0.3275911f, az, 1.0f));
    float p = __builtin_fmaf(t, 1.061405429f, -1.453152027f);
    p = __builtin_fmaf(t, p,  1.421413741f);
    p = __builtin_fmaf(t, p, -0.284496736f);
    p = __builtin_fmaf(t, p,  0.254829592f);
    p = p * t;
    const float e    = __expf(-az * az);                 // v_exp_f32 path
    const float erfa = __builtin_fmaf(-p, e, 1.0f);
    const float erfz = __builtin_copysignf(erfa, z);
    return 0.5f * x * (1.0f + erfz);
}

// Within-wave LDS store->load ordering (DS ops are in-order per wave on CDNA5;
// the split-counter wait also acts as a compiler memory fence).
__device__ __forceinline__ void lds_wave_fence() {
    asm volatile("s_wait_dscnt 0" ::: "memory");
}

// One 16-row tile through the 3-layer MLP, all layers on WMMA.
// af:   A fragment of input features (K zero-padded to 32; built in registers)
// w0h:  LDS [64 out][8]  f16, K zero-padded          b0: [64] f32
// w1h:  LDS [64 out][66] f16 (padded stride)         b1: [64] f32
// w2h:  LDS [16 out][66] f16, rows 5..15 zero        b2: [5]  f32
// acth: per-wave LDS [16][66] f16 activation buffer (reused in place)
// out5: LDS [16][5] f32 (all 16 rows written; caller masks invalid rows)
__device__ __forceinline__ void mlp_pass(
    int lane, Frag af,
    const _Float16* __restrict__ w0h, const float* __restrict__ b0,
    const _Float16* __restrict__ w1h, const float* __restrict__ b1,
    const _Float16* __restrict__ w2h, const float* __restrict__ b2,
    _Float16* __restrict__ acth, float* __restrict__ out5)
{
    const int lrow = lane & 15;
    const int hi   = lane >> 4;

    // ---- layer 0: [16xK] @ [Kx64], K padded to 32 -> 4 N-tiles, 1 wmma each.
    // B fragment: half-0 lanes hold K=0..15 (only K<8 nonzero); half-1 all zero.
    #pragma unroll
    for (int t = 0; t < 4; ++t) {
        const int col = t * 16 + lrow;
        Frag bf = {};
        if (hi == 0) {
            #pragma unroll
            for (int v = 0; v < 4; ++v)
                bf.p[v] = *(const h2*)(w0h + col * 8 + 2 * v);
        }
        v8f acc = {};
        acc = __builtin_amdgcn_wmma_f32_16x16x32_f16(false, af.v, false, bf.v,
                                                     (short)0, acc, false, false);
        const float bias = b0[col];
        #pragma unroll
        for (int v = 0; v < 8; ++v)
            acth[(v + 8 * hi) * 66 + col] = (_Float16)gelu_exact(acc[v] + bias);
    }
    lds_wave_fence();

    // ---- layer 1: [16x64] @ [64x64] -> 2 K-steps x 4 N-tiles ----
    Frag a0, a1;   // A fragments (register-resident before epilogue overwrites acth)
    #pragma unroll
    for (int v = 0; v < 8; ++v) {
        const int k0 = hi * 16 + 2 * v;
        a0.p[v] = *(const h2*)(acth + lrow * 66 + k0);
        a1.p[v] = *(const h2*)(acth + lrow * 66 + 32 + k0);
    }
    #pragma unroll
    for (int t = 0; t < 4; ++t) {
        const int col = t * 16 + lrow;
        Frag bf0, bf1;
        #pragma unroll
        for (int v = 0; v < 8; ++v) {
            const int kb = hi * 16 + 2 * v;
            bf0.p[v] = *(const h2*)(w1h + col * 66 + kb);
            bf1.p[v] = *(const h2*)(w1h + col * 66 + 32 + kb);
        }
        v8f acc = {};
        acc = __builtin_amdgcn_wmma_f32_16x16x32_f16(false, a0.v, false, bf0.v,
                                                     (short)0, acc, false, false);
        acc = __builtin_amdgcn_wmma_f32_16x16x32_f16(false, a1.v, false, bf1.v,
                                                     (short)0, acc, false, false);
        const float bias = b1[col];
        #pragma unroll
        for (int v = 0; v < 8; ++v)   // write layer-1 output in place (f16)
            acth[(v + 8 * hi) * 66 + col] = (_Float16)gelu_exact(acc[v] + bias);
    }
    lds_wave_fence();

    // ---- layer 2: [16x64] @ [64x16(5 used)] -> 2 K-steps, 1 N-tile ----
    Frag e0, e1, d0, d1;
    #pragma unroll
    for (int v = 0; v < 8; ++v) {
        const int k0 = hi * 16 + 2 * v;
        e0.p[v] = *(const h2*)(acth + lrow * 66 + k0);
        e1.p[v] = *(const h2*)(acth + lrow * 66 + 32 + k0);
    }
    #pragma unroll
    for (int v = 0; v < 8; ++v) {
        const int kb = hi * 16 + 2 * v;
        d0.p[v] = *(const h2*)(w2h + lrow * 66 + kb);
        d1.p[v] = *(const h2*)(w2h + lrow * 66 + 32 + kb);
    }
    v8f acc = {};
    acc = __builtin_amdgcn_wmma_f32_16x16x32_f16(false, e0.v, false, d0.v,
                                                 (short)0, acc, false, false);
    acc = __builtin_amdgcn_wmma_f32_16x16x32_f16(false, e1.v, false, d1.v,
                                                 (short)0, acc, false, false);
    if (lrow < 5) {
        const float bias = b2[lrow];
        #pragma unroll
        for (int v = 0; v < 8; ++v)
            out5[(v + 8 * hi) * 5 + lrow] = acc[v] + bias;
    }
    lds_wave_fence();
}

__global__ __launch_bounds__(288, 1)
void jastrow_kernel(const float* __restrict__ x,
                    const float* __restrict__ pw0, const float* __restrict__ pb0,
                    const float* __restrict__ pw1, const float* __restrict__ pb1,
                    const float* __restrict__ pw2, const float* __restrict__ pb2,
                    const float* __restrict__ qw0, const float* __restrict__ qb0,
                    const float* __restrict__ qw1, const float* __restrict__ qb1,
                    const float* __restrict__ qw2, const float* __restrict__ qb2,
                    const float* __restrict__ rw0, const float* __restrict__ rb0,
                    const float* __restrict__ rw1, const float* __restrict__ rb1,
                    float* __restrict__ out)
{
    // ---- LDS (~47 KB) ----
    __shared__ _Float16 s_pw1h[64 * 66], s_qw1h[64 * 66];
    __shared__ _Float16 s_pw0h[64 * 8],  s_qw0h[64 * 8];
    __shared__ _Float16 s_pw2h[16 * 66], s_qw2h[16 * 66];
    __shared__ _Float16 s_act0h[9][16 * 66];
    __shared__ float    s_pb0[64], s_pb1[64], s_pb2[8];
    __shared__ float    s_qb0[64], s_qb1[64], s_qb2[8];
    __shared__ float    s_x[32];
    __shared__ float    s_out_psi[128 * 5];
    __shared__ float    s_out_phi[16 * 5];
    __shared__ float    s_cusp[128];
    __shared__ float    s_part[40];   // psi segment partials [seg*5 + c]
    __shared__ float    s_cpart[8];   // cusp segment partials
    __shared__ float    s_mean[12];   // [0..4] phi mean, [5..9] psi mean, [10] cusp
    __shared__ float    s_h[64], s_h8[8];

    const int tid = threadIdx.x;
    const int b   = blockIdx.x;

    // ---- stage weights (f16, zero-padded into WMMA B-fragment shapes) ----
    for (int e = tid; e < 64 * 64; e += 288) {
        const int j = e >> 6, k = e & 63;
        s_pw1h[j * 66 + k] = (_Float16)pw1[e];
        s_qw1h[j * 66 + k] = (_Float16)qw1[e];
    }
    for (int e = tid; e < 64 * 8; e += 288) {
        const int j = e >> 3, k = e & 7;
        s_pw0h[e] = (k < 3) ? (_Float16)pw0[j * 3 + k] : (_Float16)0.f;
        s_qw0h[e] = (k < 6) ? (_Float16)qw0[j * 6 + k] : (_Float16)0.f;
    }
    for (int e = tid; e < 16 * 64; e += 288) {
        const int c = e >> 6, k = e & 63;
        s_pw2h[c * 66 + k] = (c < 5) ? (_Float16)pw2[c * 64 + k] : (_Float16)0.f;
        s_qw2h[c * 66 + k] = (c < 5) ? (_Float16)qw2[c * 64 + k] : (_Float16)0.f;
    }
    if (tid < 64) {
        s_pb0[tid] = pb0[tid]; s_pb1[tid] = pb1[tid];
        s_qb0[tid] = qb0[tid]; s_qb1[tid] = qb1[tid];
    }
    if (tid < 5) { s_pb2[tid] = pb2[tid]; s_qb2[tid] = qb2[tid]; }
    if (tid < 32) s_x[tid] = x[b * 32 + tid];   // 16 particles x 2 dims
    __syncthreads();

    const int wid  = tid >> 5;       // 0..8
    const int lane = tid & 31;
    const int lrow = lane & 15;

    if (wid < 8) {
        // ---- psi: this wave's 16 pair-rows (features register-resident) ----
        float f0 = 0.f, f1 = 0.f, f2 = 0.f, f3 = 0.f, f4 = 0.f, f5 = 0.f;
        const int p = wid * 16 + lrow;
        if ((lane < 16) && (p < 120)) {
            int pp = p, i = 0, rem = 15;          // triu(N=16, k=1) pair index
            while (pp >= rem) { pp -= rem; --rem; ++i; }
            const int j = i + 1 + pp;
            const float dx = s_x[i * 2 + 0] - s_x[j * 2 + 0];
            const float dy = s_x[i * 2 + 1] - s_x[j * 2 + 1];
            const float r  = sqrtf(dx * dx + dy * dy + 1e-12f);
            f0 = log1pf(r);
            f1 = r * __builtin_amdgcn_rcpf(1.0f + r);
            f2 = __expf(-r * r);
            f3 = __expf(-0.5f * r);
            f4 = __expf(-r);
            f5 = f4 * f4;                          // exp(-2r)
            s_cusp[p] = r * f4;                    // r * exp(-r)
        }
        Frag af = {};  // A fragment: lanes 0-15 carry K=0..7 = features; rest 0
        h2 t0 = { (_Float16)f0, (_Float16)f1 }; af.p[0] = t0;
        h2 t1 = { (_Float16)f2, (_Float16)f3 }; af.p[1] = t1;
        h2 t2 = { (_Float16)f4, (_Float16)f5 }; af.p[2] = t2;
        mlp_pass(lane, af, s_qw0h, s_qb0, s_qw1h, s_qb1, s_qw2h, s_qb2,
                 s_act0h[wid], &s_out_psi[wid * 16 * 5]);
    } else {
        // ---- phi: wave 8, rows = the 16 particles (OMEGA = 1 -> xs = x) ----
        float g0 = 0.f, g1 = 0.f, g2 = 0.f;
        if (lane < 16) {
            g0 = s_x[lrow * 2 + 0];
            g1 = s_x[lrow * 2 + 1];
            g2 = g0 * g0 + g1 * g1;
        }
        Frag ap = {};
        h2 u0 = { (_Float16)g0, (_Float16)g1 }; ap.p[0] = u0;
        h2 u1 = { (_Float16)g2, (_Float16)0.f }; ap.p[1] = u1;
        mlp_pass(lane, ap, s_pw0h, s_pb0, s_pw1h, s_pb1, s_pw2h, s_pb2,
                 s_act0h[8], s_out_phi);
    }
    __syncthreads();

    // ---- deterministic two-level reductions (fixed order -> replay-stable) ----
    if (tid < 40) {                       // psi: 8 segments x 5 cols, 15 rows each
        const int seg = tid / 5, c = tid % 5;
        float s = 0.f;
        for (int r = 0; r < 15; ++r) s += s_out_psi[(seg * 15 + r) * 5 + c];
        s_part[seg * 5 + c] = s;
    } else if (tid < 48) {                // cusp: 8 segments of 15
        const int seg = tid - 40;
        float s = 0.f;
        for (int r = 0; r < 15; ++r) s += s_cusp[seg * 15 + r];
        s_cpart[seg] = s;
    } else if (tid < 53) {                // phi: 5 cols, 16 rows each
        const int c = tid - 48;
        float s = 0.f;
        for (int r = 0; r < 16; ++r) s += s_out_phi[r * 5 + c];
        s_mean[c] = s * (1.0f / 16.0f);
    }
    __syncthreads();
    if (tid < 5) {
        float s = 0.f;
        #pragma unroll
        for (int g = 0; g < 8; ++g) s += s_part[g * 5 + tid];
        s_mean[5 + tid] = s * (1.0f / 120.0f);
    } else if (tid == 5) {
        float s = 0.f;
        #pragma unroll
        for (int g = 0; g < 8; ++g) s += s_cpart[g];
        s_mean[10] = s;
    }
    __syncthreads();

    // ---- rho readout: gelu([phi,psi] @ rho_w0^T + b0) @ rho_w1^T + b1 + cusp ----
    if (tid < 64) {
        float s = rb0[tid];
        #pragma unroll
        for (int k = 0; k < 10; ++k) s += s_mean[k] * rw0[tid * 10 + k];
        s_h[tid] = gelu_exact(s) * rw1[tid];
    }
    __syncthreads();
    if (tid < 8) {
        float s = 0.f;
        #pragma unroll
        for (int j = 0; j < 8; ++j) s += s_h[tid * 8 + j];
        s_h8[tid] = s;
    }
    __syncthreads();
    if (tid == 0) {
        float f = rb1[0];
        #pragma unroll
        for (int j = 0; j < 8; ++j) f += s_h8[j];
        out[b] = f + s_mean[10];   // CUSP_GAMMA = 1/(D-1) = 1
    }
}

extern "C" void kernel_launch(void* const* d_in, const int* in_sizes, int n_in,
                              void* d_out, int out_size, void* d_ws, size_t ws_size,
                              hipStream_t stream) {
    (void)in_sizes; (void)n_in; (void)d_ws; (void)ws_size;
    const float* xin = (const float*)d_in[0];
    const float* pw0 = (const float*)d_in[1];
    const float* pb0 = (const float*)d_in[2];
    const float* pw1 = (const float*)d_in[3];
    const float* pb1 = (const float*)d_in[4];
    const float* pw2 = (const float*)d_in[5];
    const float* pb2 = (const float*)d_in[6];
    const float* qw0 = (const float*)d_in[7];
    const float* qb0 = (const float*)d_in[8];
    const float* qw1 = (const float*)d_in[9];
    const float* qb1 = (const float*)d_in[10];
    const float* qw2 = (const float*)d_in[11];
    const float* qb2 = (const float*)d_in[12];
    const float* rw0 = (const float*)d_in[13];
    const float* rb0 = (const float*)d_in[14];
    const float* rw1 = (const float*)d_in[15];
    const float* rb1 = (const float*)d_in[16];
    float* out = (float*)d_out;

    jastrow_kernel<<<dim3((unsigned)out_size), dim3(288), 0, stream>>>(
        xin, pw0, pb0, pw1, pb1, pw2, pb2,
        qw0, qb0, qw1, qb1, qw2, qb2,
        rw0, rb0, rw1, rb1, out);
}
